// LayoutLMv2Encoder_10977936409154
// MI455X (gfx1250) — compile-verified, matrix operand order checked by reading, not run on hardware
//
#include <hip/hip_runtime.h>
#include <math.h>

#define Bn 8
#define Sn 512
#define HDn 768
#define NHn 12
#define Ln 12
#define FFn 3072
#define DHn 64
#define BSn (Bn * Sn)

typedef __bf16 bf16_t;
typedef bf16_t v16bf __attribute__((ext_vector_type(16)));
typedef float v8f __attribute__((ext_vector_type(8)));
typedef unsigned int u32x4 __attribute__((ext_vector_type(4)));

__device__ __forceinline__ unsigned short f2bf(float f) {
    unsigned int u = __float_as_uint(f);
    u += 0x7FFFu + ((u >> 16) & 1u);   // round-to-nearest-even
    return (unsigned short)(u >> 16);
}

// Async copy of 32 contiguous bytes global -> LDS (CDNA5 TDM-lite path).
// INST_OFFSET is added to BOTH the LDS and global address (ISA 10.x async ops).
__device__ __forceinline__ void async_copy32(const unsigned short* gsrc, unsigned short* ldst) {
    unsigned lds = (unsigned)(unsigned long long)(void*)ldst;  // low 32 bits = LDS offset
    asm volatile("global_load_async_to_lds_b128 %0, %1, off\n\t"
                 "global_load_async_to_lds_b128 %0, %1, off offset:16"
                 :: "v"(lds), "v"(gsrc) : "memory");
}

// ---------------------------------------------------------------------------
// Generic batched WMMA GEMM:  D = A(bf16) x B(bf16) [+ colBias] [+ Cadd]
//   A : M x K row-major (per batch, stride sAb)
//   BT: N x K row-major (B TRANSPOSED; per batch, stride sBb)
//   Cadd: M x N fp32 (per batch, stride sCb) -- residual / attention bias
//   D: fp32, row stride ldD, batch offset = (bz/divD)*sDo + (bz%divD)*sDi
// Block tile BM x BN, K-step 32, double-buffered async global->LDS staging.
// Each wave computes 32x64: 2 A-frags x 4 B-frags = 8 WMMAs per K-step.
// ---------------------------------------------------------------------------
template <int BM, int BN>
__global__ __launch_bounds__((BM / 32) * (BN / 64) * 32) void wmma_gemm_kernel(
    const unsigned short* __restrict__ A, const unsigned short* __restrict__ BT,
    const float* __restrict__ colBias, const float* __restrict__ Cadd,
    float* __restrict__ D,
    int M, int N, int K,
    long sAb, long sBb, long sCb,
    int divD, long sDo, long sDi, int ldD)
{
    constexpr int WAVES_N = BN / 64;
    constexpr int T = (BM / 32) * WAVES_N * 32;
    constexpr int AITER = (BM * 2) / T;        // 32B chunks per thread (A tile)
    constexpr int BITER = (BN * 2) / T;        // 32B chunks per thread (B tile)
    constexpr int PER_TILE = (AITER + BITER) * 2;  // async instrs per thread per tile

    __shared__ __align__(16) unsigned short lA[2][BM][32];  // [buf][m][k]
    __shared__ __align__(16) unsigned short lB[2][BN][32];  // [buf][n][k]

    const int bz = blockIdx.z;
    const unsigned short* Ab = A + (long)bz * sAb;
    const unsigned short* Bb = BT + (long)bz * sBb;
    const float* Cb = Cadd ? (Cadd + (long)bz * sCb) : nullptr;
    float* Db = D + (long)(bz / divD) * sDo + (long)(bz % divD) * sDi;

    const int rowBase = blockIdx.y * BM;
    const int colBase = blockIdx.x * BN;
    const int tid  = threadIdx.x;
    const int wave = tid >> 5;
    const int lane = tid & 31;
    const int half = lane >> 4;
    const int l16  = lane & 15;
    const int wr   = wave / WAVES_N;   // 32 rows per wave-row
    const int wc   = wave % WAVES_N;   // 64 cols per wave-col

    v8f acc[2][4] = {};
    union Frag { u32x4 q[2]; v16bf v; };

    // stage tile at k0 into LDS buffer `buf` via async copies
    auto stage = [&](int k0, int buf) {
        #pragma unroll
        for (int it = 0; it < AITER; ++it) {
            int c = tid + it * T;
            int r = c >> 1, off = (c & 1) * 16;
            const unsigned short* g = Ab + (long)(rowBase + r) * K + k0 + off;
            if (k0 + 64 < K) __builtin_prefetch(g + 64, 0, 1);
            async_copy32(g, &lA[buf][r][off]);
        }
        #pragma unroll
        for (int it = 0; it < BITER; ++it) {
            int c = tid + it * T;
            int r = c >> 1, off = (c & 1) * 16;
            const unsigned short* g = Bb + (long)(colBase + r) * K + k0 + off;
            if (k0 + 64 < K) __builtin_prefetch(g + 64, 0, 1);
            async_copy32(g, &lB[buf][r][off]);
        }
    };

    const int nt = K / 32;
    stage(0, 0);

    for (int t = 0; t < nt; ++t) {
        const int cur = t & 1;
        const bool more = (t + 1 < nt);
        if (more) stage((t + 1) * 32, 1 - cur);

        if (more) asm volatile("s_wait_asynccnt %0" :: "n"(PER_TILE) : "memory");
        else      asm volatile("s_wait_asynccnt 0" ::: "memory");
        __syncthreads();

        Frag a[2], b[4];
        #pragma unroll
        for (int i = 0; i < 2; ++i) {
            const unsigned short* ap = &lA[cur][wr * 32 + i * 16 + l16][0];
            a[i].q[0] = *(const u32x4*)(ap + half * 8);
            a[i].q[1] = *(const u32x4*)(ap + 16 + half * 8);
        }
        #pragma unroll
        for (int j = 0; j < 4; ++j) {
            const unsigned short* bp = &lB[cur][wc * 64 + j * 16 + l16][0];
            b[j].q[0] = *(const u32x4*)(bp + half * 8);
            b[j].q[1] = *(const u32x4*)(bp + 16 + half * 8);
        }
        #pragma unroll
        for (int i = 0; i < 2; ++i)
            #pragma unroll
            for (int j = 0; j < 4; ++j)
                acc[i][j] = __builtin_amdgcn_wmma_f32_16x16x32_bf16(
                    false, a[i].v, false, b[j].v, (short)0, acc[i][j], false, false);
        __syncthreads();
    }

    // --- epilogue: lane l -> col l&15, rows half*8 .. half*8+7 ---
    const int mwave = rowBase + wr * 32 + half * 8;
    const int cwave = colBase + wc * 64;
    if (Cb != nullptr) {
        #pragma unroll
        for (int i = 0; i < 2; ++i)
            #pragma unroll
            for (int j = 0; j < 4; ++j) {
                int col = cwave + j * 16 + l16;
                float cb = colBias ? colBias[col] : 0.0f;
                #pragma unroll
                for (int r = 0; r < 8; ++r) {
                    long row = mwave + i * 16 + r;
                    Db[row * ldD + col] = acc[i][j][r] + cb + Cb[row * N + col];
                }
            }
    } else {
        #pragma unroll
        for (int i = 0; i < 2; ++i)
            #pragma unroll
            for (int j = 0; j < 4; ++j) {
                int col = cwave + j * 16 + l16;
                float cb = colBias ? colBias[col] : 0.0f;
                #pragma unroll
                for (int r = 0; r < 8; ++r) {
                    long row = mwave + i * 16 + r;
                    Db[row * ldD + col] = acc[i][j][r] + cb;
                }
            }
    }
}

// ---------------------------------------------------------------------------
// Relative-position attention bias (layer invariant): bias[b][h][s][t]
// ---------------------------------------------------------------------------
__device__ __forceinline__ int rel_bucket(int rel, int nbuckets, int maxdist) {
    int nb = nbuckets / 2;
    int ret = (rel > 0) ? nb : 0;
    int n = rel < 0 ? -rel : rel;
    int max_exact = nb / 2;
    float scale = (float)(nb - max_exact) / logf((float)maxdist / (float)max_exact);
    int m = n < 1 ? 1 : n;
    int val = max_exact + (int)(logf((float)m / (float)max_exact) * scale);
    val = val < (nb - 1) ? val : (nb - 1);
    return ret + ((n < max_exact) ? n : val);
}

__global__ void bias_kernel(const int* __restrict__ pos, const int* __restrict__ bbox,
                            const float* __restrict__ rpw, const float* __restrict__ rxw,
                            const float* __restrict__ ryw, float* __restrict__ out)
{
    long idx = (long)blockIdx.x * blockDim.x + threadIdx.x;
    long total = (long)Bn * Sn * Sn;
    if (idx >= total) return;
    int t = (int)(idx % Sn);
    long r2 = idx / Sn;
    int s = (int)(r2 % Sn);
    int b = (int)(r2 / Sn);

    int bk = rel_bucket(pos[b * Sn + t] - pos[b * Sn + s], 32, 128);
    int bx = rel_bucket(bbox[((long)b * Sn + t) * 4 + 0] - bbox[((long)b * Sn + s) * 4 + 0], 64, 256);
    int by = rel_bucket(bbox[((long)b * Sn + t) * 4 + 3] - bbox[((long)b * Sn + s) * 4 + 3], 64, 256);

    #pragma unroll
    for (int h = 0; h < NHn; ++h) {
        out[(((long)b * NHn + h) * Sn + s) * Sn + t] =
            rpw[bk * NHn + h] + rxw[bx * NHn + h] + ryw[by * NHn + h];
    }
}

// ---------------------------------------------------------------------------
// fp32 -> bf16 cast (optional fp32 copy-through)
// ---------------------------------------------------------------------------
__global__ void cast_kernel(const float* __restrict__ src, unsigned short* __restrict__ dst,
                            float* __restrict__ copy, long n)
{
    long i = (long)blockIdx.x * blockDim.x + threadIdx.x;
    long stride = (long)gridDim.x * blockDim.x;
    for (; i < n; i += stride) {
        float v = src[i];
        dst[i] = f2bf(v);
        if (copy) copy[i] = v;
    }
}

// ---------------------------------------------------------------------------
// fp32 [layers][Kd][Nd] -> bf16 TRANSPOSED [layers][Nd][Kd]
// ---------------------------------------------------------------------------
__global__ void cast_transpose_kernel(const float* __restrict__ src, unsigned short* __restrict__ dst,
                                      int Kd, int Nd, long total)
{
    long idx = (long)blockIdx.x * blockDim.x + threadIdx.x;
    long stride = (long)gridDim.x * blockDim.x;
    long plane = (long)Kd * Nd;
    for (; idx < total; idx += stride) {
        long l = idx / plane;
        long rem = idx - l * plane;
        int n = (int)(rem / Kd);
        int k = (int)(rem - (long)n * Kd);
        dst[idx] = f2bf(src[l * plane + (long)k * Nd + n]);
    }
}

// ---------------------------------------------------------------------------
// Split QKV -> q (scaled), k (natural = B^T for scores), v^T (B^T for ctx)
// ---------------------------------------------------------------------------
__global__ void qkv_split_kernel(const float* __restrict__ qkv,
                                 const float* __restrict__ qb, const float* __restrict__ vb,
                                 unsigned short* __restrict__ q, unsigned short* __restrict__ k,
                                 unsigned short* __restrict__ vT)
{
    long idx = (long)blockIdx.x * blockDim.x + threadIdx.x;
    long total = (long)BSn * HDn;
    if (idx >= total) return;
    int c = (int)(idx % HDn);
    long bs = idx / HDn;
    int s = (int)(bs % Sn);
    int b = (int)(bs / Sn);
    int h = c / DHn, d = c % DHn;

    const float* row = qkv + bs * (3 * HDn);
    float qv = (row[c] + qb[c]) * 0.125f;           // 1/sqrt(64)
    float kv = row[HDn + c];
    float vv = row[2 * HDn + c] + vb[c];

    long hb = ((long)b * NHn + h);
    q [(hb * Sn + s) * DHn + d] = f2bf(qv);
    k [(hb * Sn + s) * DHn + d] = f2bf(kv);
    vT[(hb * DHn + d) * Sn + s] = f2bf(vv);
}

// ---------------------------------------------------------------------------
// Row softmax over S=512 with boolean mask; output bf16 probs
// ---------------------------------------------------------------------------
__global__ __launch_bounds__(128) void softmax_kernel(const float* __restrict__ scores,
                                                      const unsigned char* __restrict__ mask,
                                                      unsigned short* __restrict__ probs)
{
    __shared__ float red[128];
    long row = blockIdx.x;
    int b = (int)(row / ((long)NHn * Sn));
    const float* src = scores + row * Sn;
    const unsigned char* m = mask + (long)b * Sn;

    float v[4];
    float mx = -INFINITY;
    #pragma unroll
    for (int i = 0; i < 4; ++i) {
        int t = threadIdx.x + i * 128;
        float x = m[t] ? -INFINITY : src[t];
        v[i] = x;
        mx = fmaxf(mx, x);
    }
    red[threadIdx.x] = mx; __syncthreads();
    for (int o = 64; o > 0; o >>= 1) {
        if (threadIdx.x < o) red[threadIdx.x] = fmaxf(red[threadIdx.x], red[threadIdx.x + o]);
        __syncthreads();
    }
    mx = red[0]; __syncthreads();

    float sum = 0.f;
    #pragma unroll
    for (int i = 0; i < 4; ++i) { v[i] = __expf(v[i] - mx); sum += v[i]; }
    red[threadIdx.x] = sum; __syncthreads();
    for (int o = 64; o > 0; o >>= 1) {
        if (threadIdx.x < o) red[threadIdx.x] += red[threadIdx.x + o];
        __syncthreads();
    }
    float inv = 1.0f / red[0];
    #pragma unroll
    for (int i = 0; i < 4; ++i)
        probs[row * Sn + threadIdx.x + i * 128] = f2bf(v[i] * inv);
}

// ---------------------------------------------------------------------------
// Row LayerNorm over HD=768 (bias+residual already added by GEMM epilogue)
// ---------------------------------------------------------------------------
__global__ __launch_bounds__(256) void ln_kernel(const float* __restrict__ src,
                                                 const float* __restrict__ g, const float* __restrict__ b,
                                                 float* __restrict__ outf, unsigned short* __restrict__ outb)
{
    __shared__ float red[256];
    long row = blockIdx.x;
    const float* x = src + row * HDn;

    float v[3], s = 0.f, sq = 0.f;
    #pragma unroll
    for (int i = 0; i < 3; ++i) {
        int c = threadIdx.x + i * 256;
        float t = x[c];
        v[i] = t; s += t; sq += t * t;
    }
    red[threadIdx.x] = s; __syncthreads();
    for (int o = 128; o > 0; o >>= 1) {
        if (threadIdx.x < o) red[threadIdx.x] += red[threadIdx.x + o];
        __syncthreads();
    }
    float mean = red[0] * (1.0f / HDn); __syncthreads();
    red[threadIdx.x] = sq; __syncthreads();
    for (int o = 128; o > 0; o >>= 1) {
        if (threadIdx.x < o) red[threadIdx.x] += red[threadIdx.x + o];
        __syncthreads();
    }
    float var = red[0] * (1.0f / HDn) - mean * mean;
    float rstd = rsqrtf(var + 1e-12f);
    #pragma unroll
    for (int i = 0; i < 3; ++i) {
        int c = threadIdx.x + i * 256;
        float y = (v[i] - mean) * rstd * g[c] + b[c];
        outf[row * HDn + c] = y;
        outb[row * HDn + c] = f2bf(y);
    }
}

// ---------------------------------------------------------------------------
// Exact GELU (bias already added in GEMM epilogue); output bf16
// ---------------------------------------------------------------------------
__global__ void gelu_kernel(const float* __restrict__ src, unsigned short* __restrict__ dst, long n)
{
    long i = (long)blockIdx.x * blockDim.x + threadIdx.x;
    long stride = (long)gridDim.x * blockDim.x;
    for (; i < n; i += stride) {
        float x = src[i];
        dst[i] = f2bf(0.5f * x * (1.0f + erff(x * 0.70710678118654752f)));
    }
}

// ---------------------------------------------------------------------------
// Host orchestration
// ---------------------------------------------------------------------------
template <int BM, int BN>
static void launch_gemm(hipStream_t st, const unsigned short* A, const unsigned short* BT,
                        const float* colBias, const float* Cadd, float* D,
                        int M, int N, int K, int batch,
                        long sAb, long sBb, long sCb,
                        int divD, long sDo, long sDi, int ldD)
{
    constexpr int T = (BM / 32) * (BN / 64) * 32;
    dim3 grid(N / BN, M / BM, batch);
    wmma_gemm_kernel<BM, BN><<<grid, T, 0, st>>>(A, BT, colBias, Cadd, D, M, N, K,
                                                 sAb, sBb, sCb, divD, sDo, sDi, ldD);
}

extern "C" void kernel_launch(void* const* d_in, const int* in_sizes, int n_in,
                              void* d_out, int out_size, void* d_ws, size_t ws_size,
                              hipStream_t stream)
{
    const float* hs   = (const float*)d_in[0];
    const unsigned char* amask = (const unsigned char*)d_in[1];
    const int* pos    = (const int*)d_in[2];
    const int* bbox   = (const int*)d_in[3];
    const float* qkvw = (const float*)d_in[4];
    const float* qb   = (const float*)d_in[5];
    const float* vb   = (const float*)d_in[6];
    const float* aow  = (const float*)d_in[7];
    const float* aob  = (const float*)d_in[8];
    const float* g1   = (const float*)d_in[9];
    const float* b1   = (const float*)d_in[10];
    const float* iw   = (const float*)d_in[11];
    const float* ib   = (const float*)d_in[12];
    const float* ow   = (const float*)d_in[13];
    const float* ob   = (const float*)d_in[14];
    const float* g2   = (const float*)d_in[15];
    const float* b2   = (const float*)d_in[16];
    const float* rpw  = (const float*)d_in[17];
    const float* rxw  = (const float*)d_in[18];
    const float* ryw  = (const float*)d_in[19];

    // ---- workspace bump allocator ----
    char* p = (char*)d_ws;
    auto alloc = [&](size_t bytes) -> void* {
        void* r = (void*)p;
        p += (bytes + 255) & ~(size_t)255;
        return r;
    };
    // weights stored TRANSPOSED (N x K) in bf16
    unsigned short* wqkvT = (unsigned short*)alloc((size_t)Ln * HDn * 3 * HDn * 2);
    unsigned short* waoT  = (unsigned short*)alloc((size_t)Ln * HDn * HDn * 2);
    unsigned short* wintT = (unsigned short*)alloc((size_t)Ln * HDn * FFn * 2);
    unsigned short* woutT = (unsigned short*)alloc((size_t)Ln * FFn * HDn * 2);
    float*          biasbuf = (float*)alloc((size_t)Bn * NHn * Sn * Sn * 4);
    float*          h       = (float*)alloc((size_t)BSn * HDn * 4);
    unsigned short* h_bf    = (unsigned short*)alloc((size_t)BSn * HDn * 2);
    float*          qkv     = (float*)alloc((size_t)BSn * 3 * HDn * 4);
    unsigned short* q_bf    = (unsigned short*)alloc((size_t)Bn * NHn * Sn * DHn * 2);
    unsigned short* k_bf    = (unsigned short*)alloc((size_t)Bn * NHn * Sn * DHn * 2);
    unsigned short* vT_bf   = (unsigned short*)alloc((size_t)Bn * NHn * Sn * DHn * 2);
    float*          scores  = (float*)alloc((size_t)Bn * NHn * Sn * Sn * 4);
    unsigned short* prob_bf = (unsigned short*)alloc((size_t)Bn * NHn * Sn * Sn * 2);
    float*          ctx     = (float*)alloc((size_t)BSn * HDn * 4);
    unsigned short* ctx_bf  = (unsigned short*)alloc((size_t)BSn * HDn * 2);
    float*          t0      = (float*)alloc((size_t)BSn * FFn * 4);
    float*          attn    = (float*)alloc((size_t)BSn * HDn * 4);
    unsigned short* attn_bf = (unsigned short*)alloc((size_t)BSn * HDn * 2);
    unsigned short* ffn_bf  = (unsigned short*)alloc((size_t)BSn * FFn * 2);

    auto blocks = [](long n, int t) { return (unsigned)((n + t - 1) / t); };

    // ---- one-time setup: transposed weight casts + attention bias ----
    {
        long n;
        n = (long)Ln * HDn * 3 * HDn;
        cast_transpose_kernel<<<blocks(n, 256), 256, 0, stream>>>(qkvw, wqkvT, HDn, 3 * HDn, n);
        n = (long)Ln * HDn * HDn;
        cast_transpose_kernel<<<blocks(n, 256), 256, 0, stream>>>(aow, waoT, HDn, HDn, n);
        n = (long)Ln * HDn * FFn;
        cast_transpose_kernel<<<blocks(n, 256), 256, 0, stream>>>(iw, wintT, HDn, FFn, n);
        n = (long)Ln * FFn * HDn;
        cast_transpose_kernel<<<blocks(n, 256), 256, 0, stream>>>(ow, woutT, FFn, HDn, n);
        n = (long)BSn * HDn;
        cast_kernel<<<blocks(n, 256), 256, 0, stream>>>(hs, h_bf, h, n);
        long nb = (long)Bn * Sn * Sn;
        bias_kernel<<<blocks(nb, 256), 256, 0, stream>>>(pos, bbox, rpw, rxw, ryw, biasbuf);
    }

    const int BH = Bn * NHn;

    for (int l = 0; l < Ln; ++l) {
        // qkv = h @ qkv_w[l]
        launch_gemm<128, 128>(stream, h_bf, wqkvT + (size_t)l * HDn * 3 * HDn,
                              nullptr, nullptr, qkv,
                              BSn, 3 * HDn, HDn, 1, 0, 0, 0, 1, 0, 0, 3 * HDn);

        long nsp = (long)BSn * HDn;
        qkv_split_kernel<<<blocks(nsp, 256), 256, 0, stream>>>(
            qkv, qb + (size_t)l * HDn, vb + (size_t)l * HDn, q_bf, k_bf, vT_bf);

        // scores = q @ k^T + bias : BT = k (natural [s][d] layout)
        launch_gemm<128, 128>(stream, q_bf, k_bf, nullptr, biasbuf, scores,
                              Sn, Sn, DHn, BH,
                              (long)Sn * DHn, (long)Sn * DHn, (long)Sn * Sn,
                              1, (long)Sn * Sn, 0, Sn);

        softmax_kernel<<<(unsigned)((long)BH * Sn), 128, 0, stream>>>(scores, amask, prob_bf);

        // ctx = probs @ v : BT = v^T [d][s]; written merged as [B][S][HD]
        launch_gemm<128, 64>(stream, prob_bf, vT_bf, nullptr, nullptr, ctx,
                             Sn, DHn, Sn, BH,
                             (long)Sn * Sn, (long)DHn * Sn, 0,
                             NHn, (long)Sn * HDn, (long)DHn, HDn);

        cast_kernel<<<blocks(nsp, 256), 256, 0, stream>>>(ctx, ctx_bf, nullptr, nsp);

        // t0 = ctx @ attn_out_w + attn_out_b + h (residual)
        launch_gemm<128, 128>(stream, ctx_bf, waoT + (size_t)l * HDn * HDn,
                              aob + (size_t)l * HDn, h, t0,
                              BSn, HDn, HDn, 1, 0, 0, 0, 1, 0, 0, HDn);

        ln_kernel<<<BSn, 256, 0, stream>>>(t0, g1 + (size_t)l * HDn, b1 + (size_t)l * HDn, attn, attn_bf);

        // t0 = attn @ inter_w + inter_b
        launch_gemm<128, 128>(stream, attn_bf, wintT + (size_t)l * HDn * FFn,
                              ib + (size_t)l * FFn, nullptr, t0,
                              BSn, FFn, HDn, 1, 0, 0, 0, 1, 0, 0, FFn);

        long nff = (long)BSn * FFn;
        gelu_kernel<<<blocks(nff, 256), 256, 0, stream>>>(t0, ffn_bf, nff);

        // t0 = gelu @ out_w + out_b + attn (residual)
        launch_gemm<128, 128>(stream, ffn_bf, woutT + (size_t)l * FFn * HDn,
                              ob + (size_t)l * HDn, attn, t0,
                              BSn, HDn, FFn, 1, 0, 0, 0, 1, 0, 0, HDn);

        ln_kernel<<<BSn, 256, 0, stream>>>(t0, g2 + (size_t)l * HDn, b2 + (size_t)l * HDn, h, h_bf);
    }

    hipMemcpyAsync(d_out, h, (size_t)BSn * HDn * sizeof(float),
                   hipMemcpyDeviceToDevice, stream);
}
